// Geometric_21784074126013
// MI455X (gfx1250) — matrix operand order repaired and checked by reading, not compiled, not run on hardware
//
#include <hip/hip_runtime.h>
#include <hip/hip_bf16.h>
#include <cstdint>
#include <cstddef>

#define HEADS 16
#define NEG_SLOPE 0.2f

typedef __bf16 bf16_t;
typedef __attribute__((ext_vector_type(16))) __bf16 v16bf;
typedef __attribute__((ext_vector_type(8)))  __bf16 v8bf;
typedef __attribute__((ext_vector_type(8)))  float  v8f;

__device__ __forceinline__ float lrelu(float x) { return x > 0.f ? x : NEG_SLOPE * x; }

// ordered float atomic max via sign-split int/uint atomics (init must be -inf)
__device__ __forceinline__ float atomicMaxF(float* addr, float v) {
  if (v >= 0.f)
    return __int_as_float(atomicMax((int*)addr, __float_as_int(v)));
  else
    return __uint_as_float(atomicMin((unsigned int*)addr, __float_as_uint(v)));
}

// ---------------- conversion kernels ----------------

// activations: [M,K] f32 (optional relu) -> [M,Kp] bf16, zero-padded K
__global__ void k_cvt_act(const float* __restrict__ x, bf16_t* __restrict__ xb,
                          int M, int K, int Kp, int relu) {
  long long idx = (long long)blockIdx.x * blockDim.x + threadIdx.x;
  if (idx >= (long long)M * Kp) return;
  int m = (int)(idx / Kp), k = (int)(idx % Kp);
  float v = 0.f;
  if (k < K) { v = x[(size_t)m * K + k]; if (relu && v < 0.f) v = 0.f; }
  xb[idx] = (bf16_t)v;
}

// weights: W [K,Nr] f32 row-major -> Wt [Np,Kp] bf16 (transposed, zero-padded)
__global__ void k_cvt_wt(const float* __restrict__ W, bf16_t* __restrict__ wt,
                         int K, int Nr, int Kp, int Np) {
  long long idx = (long long)blockIdx.x * blockDim.x + threadIdx.x;
  if (idx >= (long long)Np * Kp) return;
  int n = (int)(idx / Kp), k = (int)(idx % Kp);
  float v = (n < Nr && k < K) ? W[(size_t)k * Nr + n] : 0.f;
  wt[idx] = (bf16_t)v;
}

// ---------------- WMMA GEMM: C[M,Nreal] = A[M,Kp](bf16) @ Bt[Np,Kp]^T (+bias) ----------------
__global__ void k_gemm_wmma(const bf16_t* __restrict__ A, const bf16_t* __restrict__ Bt,
                            float* __restrict__ C, const float* __restrict__ bias,
                            int M, int Nreal, int Kp, int Np) {
  const int lane = threadIdx.x & 31;
  const int wave = threadIdx.x >> 5;
  const int ntn = Np >> 4;
  const int Mt = (M + 15) >> 4;
  const long long ntiles = (long long)Mt * ntn;
  const long long tile = (long long)blockIdx.x * (blockDim.x >> 5) + wave;
  if (tile >= ntiles) return;                       // wave-uniform
  const int mt = (int)(tile / ntn), nt = (int)(tile % ntn);
  const int m0 = mt << 4, n0 = nt << 4;
  const int half = lane >> 4;                       // 0 | 1
  const int l15  = lane & 15;

  int arow = m0 + l15;
  int arow_c = arow < M ? arow : M - 1;             // clamp; OOB rows never stored
  const bf16_t* ap = A + (size_t)arow_c * Kp + (half ? 16 : 0);
  const bf16_t* bp = Bt + (size_t)(n0 + l15) * Kp + (half ? 16 : 0);

  const int ccol = n0 + l15;
  float binit = 0.f;
  if (bias && ccol < Nreal) binit = bias[ccol];
  v8f acc;
#pragma unroll
  for (int r = 0; r < 8; ++r) acc[r] = binit;

  for (int k0 = 0; k0 < Kp; k0 += 32) {
    // A fragment: lane row m0+l15, K = k0+16h .. k0+16h+15
    v8bf alo = *(const v8bf*)(ap + k0);
    v8bf ahi = *(const v8bf*)(ap + k0 + 8);
    v16bf a;
#pragma unroll
    for (int i = 0; i < 8; ++i) { a[i] = alo[i]; a[i + 8] = ahi[i]; }
    // B fragment: lane col n0+l15, K = k0+16h .. k0+16h+15 (contiguous in Wt)
    v16bf b = *(const v16bf*)(bp + k0);
    acc = __builtin_amdgcn_wmma_f32_16x16x32_bf16(false, a, false, b,
                                                  (short)0, acc, false, false);
  }

  const int crow0 = m0 + (half ? 8 : 0);
  if (ccol < Nreal) {
#pragma unroll
    for (int r = 0; r < 8; ++r) {
      int crow = crow0 + r;
      if (crow < M) C[(size_t)crow * Nreal + ccol] = acc[r];
    }
  }
}

// ---------------- attention coefficients ----------------
__global__ void k_attn(const float* __restrict__ h, const float* __restrict__ a_s,
                       const float* __restrict__ a_d, float* __restrict__ es,
                       float* __restrict__ ed, int Nn, int C) {
  long long idx = (long long)blockIdx.x * blockDim.x + threadIdx.x;
  if (idx >= (long long)Nn * HEADS) return;
  int n = (int)(idx / HEADS), hh = (int)(idx % HEADS);
  const float* hp = h + (size_t)n * HEADS * C + (size_t)hh * C;
  float ss = 0.f, sd = 0.f;
  for (int c = 0; c < C; ++c) { float v = hp[c]; ss += v * a_s[hh * C + c]; sd += v * a_d[hh * C + c]; }
  es[idx] = ss; ed[idx] = sd;
}

__global__ void k_init_out(float* __restrict__ out, const float* __restrict__ bias,
                           int Nn, int HC) {
  long long idx = (long long)blockIdx.x * blockDim.x + threadIdx.x;
  if (idx >= (long long)Nn * HC) return;
  out[idx] = bias[idx % HC];
}

__global__ void k_init_md(float* __restrict__ m, float* __restrict__ den, int n) {
  int idx = blockIdx.x * blockDim.x + threadIdx.x;
  if (idx >= n) return;
  m[idx] = -INFINITY; den[idx] = 0.f;
}

__device__ __forceinline__ void edge_sd(const long long* __restrict__ ei, int E,
                                        int ee, int& s, int& d) {
  if (ee < E) { s = (int)ei[ee]; d = (int)ei[(size_t)E + ee]; }
  else { s = d = ee - E; }                         // implicit self-loop
}

__global__ void k_edge_max(const long long* __restrict__ ei, int E, int Etot,
                           const float* __restrict__ es, const float* __restrict__ ed,
                           float* __restrict__ m) {
  long long idx = (long long)blockIdx.x * blockDim.x + threadIdx.x;
  if (idx >= (long long)Etot * HEADS) return;
  int ee = (int)(idx / HEADS), hh = (int)(idx & (HEADS - 1));
  int s, d; edge_sd(ei, E, ee, s, d);
  float e = lrelu(es[(size_t)s * HEADS + hh] + ed[(size_t)d * HEADS + hh]);
  atomicMaxF(&m[(size_t)d * HEADS + hh], e);
}

__global__ void k_edge_expsum(const long long* __restrict__ ei, int E, int Etot,
                              const float* __restrict__ es, const float* __restrict__ ed,
                              const float* __restrict__ m, float* __restrict__ den) {
  long long idx = (long long)blockIdx.x * blockDim.x + threadIdx.x;
  if (idx >= (long long)Etot * HEADS) return;
  int ee = (int)(idx / HEADS), hh = (int)(idx & (HEADS - 1));
  int s, d; edge_sd(ei, E, ee, s, d);
  float e = lrelu(es[(size_t)s * HEADS + hh] + ed[(size_t)d * HEADS + hh]);
  float ex = __expf(e - m[(size_t)d * HEADS + hh]);
  atomicAdd(&den[(size_t)d * HEADS + hh], ex);
}

__global__ void k_edge_aggr(const long long* __restrict__ ei, int E, int Etot,
                            const float* __restrict__ es, const float* __restrict__ ed,
                            const float* __restrict__ m, const float* __restrict__ den,
                            const float* __restrict__ h, float* __restrict__ out, int C) {
  const int HC = HEADS * C;
  const int tpe = HC >> 2;                         // threads per edge (4 ch each)
  long long gid = (long long)blockIdx.x * blockDim.x + threadIdx.x;
  int ee = (int)(gid / tpe);
  if (ee >= Etot) return;
  int t = (int)(gid % tpe);
  int c0 = t << 2;
  int hh = c0 / C;
  int s, d; edge_sd(ei, E, ee, s, d);
  float e = lrelu(es[(size_t)s * HEADS + hh] + ed[(size_t)d * HEADS + hh]);
  float alpha = __expf(e - m[(size_t)d * HEADS + hh]) /
                (den[(size_t)d * HEADS + hh] + 1e-16f);
  float4 hv = *(const float4*)(h + (size_t)s * HC + c0);
  float* op = out + (size_t)d * HC + c0;
  atomicAdd(op + 0, hv.x * alpha);
  atomicAdd(op + 1, hv.y * alpha);
  atomicAdd(op + 2, hv.z * alpha);
  atomicAdd(op + 3, hv.w * alpha);
}

// ---------------- host ----------------
extern "C" void kernel_launch(void* const* d_in, const int* in_sizes, int n_in,
                              void* d_out, int out_size, void* d_ws, size_t ws_size,
                              hipStream_t stream) {
  const float* x = (const float*)d_in[0];
  const long long* ei = (const long long*)d_in[1];
  const float* W[4]  = {(const float*)d_in[2], (const float*)d_in[6],
                        (const float*)d_in[10], (const float*)d_in[14]};
  const float* As[4] = {(const float*)d_in[3], (const float*)d_in[7],
                        (const float*)d_in[11], (const float*)d_in[15]};
  const float* Ad[4] = {(const float*)d_in[4], (const float*)d_in[8],
                        (const float*)d_in[12], (const float*)d_in[16]};
  const float* Bs[4] = {(const float*)d_in[5], (const float*)d_in[9],
                        (const float*)d_in[13], (const float*)d_in[17]};
  const float* fcw = (const float*)d_in[18];
  const float* fcb = (const float*)d_in[19];
  const float* ow  = (const float*)d_in[20];
  const float* ob  = (const float*)d_in[21];
  (void)n_in; (void)out_size; (void)ws_size;

  const int Nn = in_sizes[0] / 13;
  const int E  = in_sizes[1] / 2;
  const int Etot = E + Nn;

  size_t off = 0;
  auto carve = [&](size_t bytes) -> void* {
    void* p = (void*)((char*)d_ws + off);
    off += (bytes + 255) & ~(size_t)255;
    return p;
  };
  float*  hbuf   = (float*)carve((size_t)Nn * 256 * sizeof(float));
  float*  outbuf = (float*)carve((size_t)Nn * 256 * sizeof(float));
  bf16_t* xbf    = (bf16_t*)carve((size_t)Nn * 256 * sizeof(bf16_t));
  bf16_t* wtbf   = (bf16_t*)carve((size_t)256 * 256 * sizeof(bf16_t));
  float*  es  = (float*)carve((size_t)Nn * HEADS * sizeof(float));
  float*  edt = (float*)carve((size_t)Nn * HEADS * sizeof(float));
  float*  mm  = (float*)carve((size_t)Nn * HEADS * sizeof(float));
  float*  dd  = (float*)carve((size_t)Nn * HEADS * sizeof(float));

  auto cdiv = [](long long a, long long b) -> int { return (int)((a + b - 1) / b); };

  auto gemm = [&](const float* src, int K, int relu_in,
                  const float* Wm, int Nout, const float* bias, float* Cdst) {
    int Kp = (K + 31) & ~31;
    int Np = (Nout + 15) & ~15;
    k_cvt_act<<<cdiv((long long)Nn * Kp, 256), 256, 0, stream>>>(src, xbf, Nn, K, Kp, relu_in);
    k_cvt_wt<<<cdiv((long long)Np * Kp, 256), 256, 0, stream>>>(Wm, wtbf, K, Nout, Kp, Np);
    long long ntiles = (long long)((Nn + 15) / 16) * (Np / 16);
    k_gemm_wmma<<<cdiv(ntiles, 8), 256, 0, stream>>>(xbf, wtbf, Cdst, bias, Nn, Nout, Kp, Np);
  };

  const int Cs[4] = {16, 16, 16, 8};
  const int Ks[4] = {13, 256, 256, 256};
  const float* lin = x;
  for (int li = 0; li < 4; ++li) {
    const int C = Cs[li], K = Ks[li], HC = HEADS * C;
    gemm(lin, K, li > 0 ? 1 : 0, W[li], HC, nullptr, hbuf);   // h = relu(prev) @ W
    k_attn<<<cdiv((long long)Nn * HEADS, 256), 256, 0, stream>>>(hbuf, As[li], Ad[li], es, edt, Nn, C);
    k_init_out<<<cdiv((long long)Nn * HC, 256), 256, 0, stream>>>(outbuf, Bs[li], Nn, HC);
    k_init_md<<<cdiv((long long)Nn * HEADS, 256), 256, 0, stream>>>(mm, dd, Nn * HEADS);
    k_edge_max<<<cdiv((long long)Etot * HEADS, 256), 256, 0, stream>>>(ei, E, Etot, es, edt, mm);
    k_edge_expsum<<<cdiv((long long)Etot * HEADS, 256), 256, 0, stream>>>(ei, E, Etot, es, edt, mm, dd);
    const int tpe = HC / 4;
    k_edge_aggr<<<cdiv((long long)Etot * tpe, 256), 256, 0, stream>>>(ei, E, Etot, es, edt, mm, dd, hbuf, outbuf, C);
    lin = outbuf;
  }
  // MLP head: [N,128] -> [N,256] -> [N,10]
  gemm(outbuf, 128, 0, fcw, 256, fcb, hbuf);
  gemm(hbuf, 256, 0, ow, 10, ob, (float*)d_out);
}
